// MinervaEnhancedLoss_62818191671798
// MI455X (gfx1250) — compile-verified, matrix operand order checked by reading, not run on hardware
//
#include <hip/hip_runtime.h>
#include <hip/hip_bf16.h>

// ---------------------------------------------------------------------------
// MinervaEnhancedLoss for MI455X (gfx1250, wave32).
// Bandwidth-bound: ~101 MB in -> 10 scalars (~4.3 us floor @ 23.3 TB/s).
// Kernel A: one block per sample, 4 pixels/thread/sweep with b128 class-plane
// loads (global_load_b128, fully coalesced), focal / argmax / IoU / copy /
// 2x2-code-diversity / sigmoid partials, deterministic shuffle+LDS reductions.
// Kernel B: one wave folds the 512-sample partial arrays with
// V_WMMA_F32_16X16X4_F32 (B = ones => D[m][n] = rowsum_m; total = sum of all
// A elements, independent of A lane layout).
// ---------------------------------------------------------------------------

typedef __attribute__((ext_vector_type(2)))  float    v2f;
typedef __attribute__((ext_vector_type(8)))  float    v8f;
typedef __attribute__((ext_vector_type(16))) _Float16 v16h;

#define N_B   512
#define N_C   10
#define HDIM  64
#define WDIM  64
#define HW    4096      // 64*64
#define NWIN  3969      // 63*63
#define BITW  313       // ceil(10000/32)
#define NQ    6         // per-sample partial quantities

__device__ __forceinline__ float wave_sum_f(float v) {
#pragma unroll
  for (int o = 16; o > 0; o >>= 1) v += __shfl_xor(v, o, 32);
  return v;
}
__device__ __forceinline__ int wave_sum_i(int v) {
#pragma unroll
  for (int o = 16; o > 0; o >>= 1) v += __shfl_xor(v, o, 32);
  return v;
}
__device__ __forceinline__ int wave_and_i(int v) {
#pragma unroll
  for (int o = 16; o > 0; o >>= 1) v &= __shfl_xor(v, o, 32);
  return v;
}

// ---------------------------------------------------------------------------
// Kernel A: per-sample statistics. grid = 512 blocks, block = 256 threads.
// ws layout: ws[q*512 + b], q = 0..5:
//   0: strategic_weight * sum_pixels((1-pt)^2 * ce)
//   1: iou_b                     (= eq_count / 4096)
//   2: exact_strict_b            (0/1)
//   3: copy_b                    (0/1)
//   4: nuniq_2x2_codes_b / 3969
//   5: sum_j sigmoid(sr[b,j])
// ---------------------------------------------------------------------------
__global__ __launch_bounds__(256) void minerva_per_sample(
    const float* __restrict__ pred,     // [512,10,64,64]
    const int*   __restrict__ targets,  // [512,64,64]
    const int*   __restrict__ inputsG,  // [512,64,64]
    const float* __restrict__ sr,       // [512,128]
    float* __restrict__ ws)             // [6*512]
{
  __shared__ unsigned char predIdxS[HW];    // per-pixel argmax class
  __shared__ unsigned int  bitmap[BITW];    // 10000-bit 2x2-code presence set
  __shared__ unsigned int  presentMask;     // per-sample target-class bitmask
  __shared__ float         wsum[8][NQ];     // per-wave partials

  const int b   = blockIdx.x;
  const int tid = threadIdx.x;

  for (int i = tid; i < BITW; i += 256) bitmap[i] = 0u;
  if (tid == 0) presentMask = 0u;
  __syncthreads();

  const float* pb = pred    + (size_t)b * N_C * HW;
  const int*   tb = targets + (size_t)b * HW;
  const int*   ib = inputsG + (size_t)b * HW;

  float focal = 0.0f;
  int   eqc = 0, alleq = 1, allcopy = 1;
  unsigned int localPresent = 0u;

  // 4 sweeps x (4 consecutive pixels per thread): b128 loads per class plane,
  // 512B per wave per class, fully coalesced. 4x fewer VMEM ops than b32.
#pragma unroll 1
  for (int i = 0; i < 4; ++i) {
    const int p4 = (i * 256 + tid) * 4;
    __builtin_prefetch(pb + p4 + 1024, 0, 1);   // global_prefetch_b8 (next sweep)

    float4 x4[N_C];
#pragma unroll
    for (int c = 0; c < N_C; ++c)
      x4[c] = *reinterpret_cast<const float4*>(pb + c * HW + p4);
    const int4 t4  = *reinterpret_cast<const int4*>(tb + p4);
    const int4 in4 = *reinterpret_cast<const int4*>(ib + p4);

    const float* xf = reinterpret_cast<const float*>(&x4[0]);  // xf[c*4 + j]
    const int*   ti = reinterpret_cast<const int*>(&t4);
    const int*   ii = reinterpret_cast<const int*>(&in4);

    uchar4 pk;
    unsigned char* pkb = reinterpret_cast<unsigned char*>(&pk);
#pragma unroll
    for (int j = 0; j < 4; ++j) {
      // argmax with first-max tie rule (matches jnp.argmax)
      float m = xf[0 * 4 + j]; int pi = 0;
#pragma unroll
      for (int c = 1; c < N_C; ++c) {
        const float xc = xf[c * 4 + j];
        if (xc > m) { m = xc; pi = c; }
      }
      float s = 0.0f;
#pragma unroll
      for (int c = 0; c < N_C; ++c) s += __expf(xf[c * 4 + j] - m);

      const int t = ti[j];
      float xt = xf[0 * 4 + j];
#pragma unroll
      for (int c = 1; c < N_C; ++c) xt = (t == c) ? xf[c * 4 + j] : xt;

      const float ce = __logf(s) - (xt - m);      // = m + lse - x_t
      const float pt = __expf(-ce);
      const float om = 1.0f - pt;
      focal += om * om * ce;

      const int eq = (pi == t);
      eqc     += eq;
      alleq   &= eq;
      allcopy &= (pi == ii[j]);
      localPresent |= (1u << t);
      pkb[j] = (unsigned char)pi;
    }
    *reinterpret_cast<uchar4*>(&predIdxS[p4]) = pk;   // one ds_store_b32
  }
  atomicOr(&presentMask, localPresent);
  __syncthreads();

  // 2x2-window distinct-code count via LDS presence bitmap (codes < 10000)
  for (int q = tid; q < NWIN; q += 256) {
    const int r = q / 63, c = q - r * 63;
    const int base = r * WDIM + c;
    const int code = (int)predIdxS[base]            * 1000 +
                     (int)predIdxS[base + 1]        * 100  +
                     (int)predIdxS[base + WDIM]     * 10   +
                     (int)predIdxS[base + WDIM + 1];
    atomicOr(&bitmap[code >> 5], 1u << (code & 31));
  }
  __syncthreads();

  int popcs = 0;
  for (int i = tid; i < BITW; i += 256) popcs += __popc(bitmap[i]);

  // sigmoid partial (threads 0..127 <=> waves 0..3, wave-uniform predicate)
  float sig = 0.0f;
  if (tid < 128) {
    const float v = sr[(size_t)b * 128 + tid];
    sig = 1.0f / (1.0f + __expf(-v));
  }

  // deterministic wave reductions
  const int wid = tid >> 5, lane = tid & 31;
  const float rf = wave_sum_f(focal);
  const int   re = wave_sum_i(eqc);
  const int   ra = wave_and_i(alleq);
  const int   rc = wave_and_i(allcopy);
  const int   rp = wave_sum_i(popcs);
  const float rs = wave_sum_f(sig);
  if (lane == 0) {
    wsum[wid][0] = rf;        wsum[wid][1] = (float)re;
    wsum[wid][2] = (float)ra; wsum[wid][3] = (float)rc;
    wsum[wid][4] = (float)rp; wsum[wid][5] = rs;
  }
  __syncthreads();

  if (tid == 0) {
    float F = 0.f, E = 0.f, P = 0.f, S = 0.f, A = 1.f, C = 1.f;
#pragma unroll
    for (int w = 0; w < 8; ++w) {
      F += wsum[w][0]; E += wsum[w][1];
      A *= wsum[w][2]; C *= wsum[w][3];
      P += wsum[w][4]; S += wsum[w][5];
    }
    const int   uniq = __popc(presentMask);
    const float wgt  = (uniq > 3) ? 1.2f : 1.0f;
    ws[0 * N_B + b] = wgt * F;
    ws[1 * N_B + b] = E * (1.0f / 4096.0f);
    ws[2 * N_B + b] = A;
    ws[3 * N_B + b] = C;
    ws[4 * N_B + b] = P * (1.0f / 3969.0f);
    ws[5 * N_B + b] = S;
  }
}

// ---------------------------------------------------------------------------
// Kernel B: one wave32. WMMA reduction of each 512-element partial array:
// with B = ones, D[m][n] = sum_k A[m][k], so the sum over all A elements is
// sum_m(rowsum_m) = (lane sum of 8 acc regs) + shfl_xor(...,16).
// ---------------------------------------------------------------------------
__global__ __launch_bounds__(32) void minerva_finalize(
    const float* __restrict__ ws, float* __restrict__ out)
{
  const int lane = threadIdx.x;
  float S[NQ];

#if __has_builtin(__builtin_amdgcn_wmma_f32_16x16x4_f32)
  v2f Bones; Bones[0] = 1.0f; Bones[1] = 1.0f;
#pragma unroll
  for (int q = 0; q < NQ; ++q) {
    v8f acc = {};
    const float* base = ws + q * N_B;
#pragma unroll
    for (int ch = 0; ch < 8; ++ch) {              // 8 chunks x 64 f32 = 512
      const float* p = base + ch * 64 + lane * 2;
      v2f Am; Am[0] = p[0]; Am[1] = p[1];
      acc = __builtin_amdgcn_wmma_f32_16x16x4_f32(
          false, Am, false, Bones, (short)0, acc, false, false);
    }
    float s = acc[0] + acc[1] + acc[2] + acc[3] +
              acc[4] + acc[5] + acc[6] + acc[7];
    s += __shfl_xor(s, 16, 32);                   // rows 0..7 + rows 8..15
    S[q] = s;
  }
#else
  // Fallback: single f16 WMMA per array (codegen-verified builtin); the
  // ones-B trick makes the total independent of the A lane layout.
  v16h Bones;
#pragma unroll
  for (int j = 0; j < 16; ++j) Bones[j] = (_Float16)1.0f;
#pragma unroll
  for (int q = 0; q < NQ; ++q) {
    const float* base = ws + q * N_B + lane * 16;
    v16h Am;
#pragma unroll
    for (int j = 0; j < 16; ++j) Am[j] = (_Float16)base[j];
    v8f acc = {};
    acc = __builtin_amdgcn_wmma_f32_16x16x32_f16(
        false, Am, false, Bones, (short)0, acc, false, false);
    float s = acc[0] + acc[1] + acc[2] + acc[3] +
              acc[4] + acc[5] + acc[6] + acc[7];
    s += __shfl_xor(s, 16, 32);
    S[q] = s;
  }
#endif

  const float focal         = S[0] * (1.0f / (512.0f * 4096.0f));
  const float iou_mean      = S[1] * (1.0f / 512.0f);
  const float combined_sum  = 0.2f * S[2] + 0.8f * S[1];
  const float combined_mean = combined_sum * (1.0f / 512.0f);
  const float transform     = (S[3] * (1.0f / 512.0f)) * 0.2f;
  const float exact_bonus   = fmaxf(-combined_mean * 5.0f, -3.0f);
  const float creativity    = (S[5] * (1.0f / (512.0f * 128.0f))) * 0.15f;
  const float diversity     = (S[4] * (1.0f / 512.0f)) * 0.02f;
  const float grid_bonus    = combined_mean * 0.05f;   // grid_size_factor==1 (4096/900>1)
  float total = focal + transform + exact_bonus
              - creativity - diversity - grid_bonus;
  const bool bad = !(total == total) || (fabsf(total) == __builtin_inff());
  if (bad) total = fminf(focal, 10.0f);

  if (lane == 0) {
    out[0] = total;
    out[1] = focal;
    out[2] = transform;
    out[3] = exact_bonus;
    out[4] = combined_sum;   // exact_count
    out[5] = combined_sum;   // combined_matches.sum()
    out[6] = iou_mean;
    out[7] = creativity;
    out[8] = diversity;
    out[9] = grid_bonus;
  }
}

extern "C" void kernel_launch(void* const* d_in, const int* in_sizes, int n_in,
                              void* d_out, int out_size, void* d_ws, size_t ws_size,
                              hipStream_t stream) {
  const float* pred    = (const float*)d_in[0];
  const int*   targets = (const int*)d_in[1];
  const int*   inputsG = (const int*)d_in[2];
  const float* sr      = (const float*)d_in[3];
  float*       ws      = (float*)d_ws;        // needs 6*512*4 = 12 KB
  float*       out     = (float*)d_out;

  minerva_per_sample<<<N_B, 256, 0, stream>>>(pred, targets, inputsG, sr, ws);
  minerva_finalize<<<1, 32, 0, stream>>>(ws, out);
}
